// FouDoRALayer_3642132267592
// MI455X (gfx1250) — matrix operand order repaired and checked by reading, not compiled, not run on hardware
//
#include <hip/hip_runtime.h>
#include <hip/hip_bf16.h>
#include <stdint.h>

// ---------------------------------------------------------------------------
// FouDoRA layer for MI455X (gfx1250, wave32).
//   out = x @ W_dora^T + bias
//   W_dora = m * (W + C) / (||W + C||_row + eps)
//   C = irfft2(pad(SCALING * (dr + i*di)))  -- 15 separable trig terms.
// GEMM runs on v_wmma_f32_16x16x32_bf16 with TDM (tensor_load_to_lds)
// double-buffered LDS staging.
// ---------------------------------------------------------------------------

typedef __bf16 bf16;
typedef bf16  v4bf  __attribute__((ext_vector_type(4)));
typedef bf16  v8bf  __attribute__((ext_vector_type(8)));
typedef bf16  v16bf __attribute__((ext_vector_type(16)));
typedef float v8f   __attribute__((ext_vector_type(8)));
typedef unsigned int u32x4 __attribute__((ext_vector_type(4)));
typedef int   i32x8 __attribute__((ext_vector_type(8)));
typedef int   i32x4 __attribute__((ext_vector_type(4)));

#define IN_DIM   4096
#define OUT_DIM  4096
#define MROWS    8192      // 4 * 2048
#define NFREQ    8
#define TWO_PI_F 6.28318530717958647692f
#define EPS_F    1e-8f

// ---------------------------------------------------------------------------
// Kernel A: per-row Fourier coefficients.
//   Z_r[v] = (S/sqrt(OUT)) * sum_u (dr+i*di)[u,v] * e^{+2pi i u r / OUT}
//   A_v(r) = 2S/4096 * Re Z' (v>=1), A_0 = S/4096 * Re Z'_0, B_v = -2S/4096 Im Z'
// with S = SCALING = 2.0  =>  2S/4096 = 1/1024.
// ---------------------------------------------------------------------------
__global__ __launch_bounds__(256)
void fou_coeffs(const float* __restrict__ dr, const float* __restrict__ di,
                float* __restrict__ Ac, float* __restrict__ Bc)
{
    int r = blockIdx.x * 256 + threadIdx.x;
    if (r >= OUT_DIM) return;
    float re[NFREQ], im[NFREQ];
#pragma unroll
    for (int v = 0; v < NFREQ; ++v) { re[v] = 0.f; im[v] = 0.f; }
    float base = (TWO_PI_F / (float)OUT_DIM) * (float)r;
#pragma unroll
    for (int u = 0; u < NFREQ; ++u) {
        float s, c;
        __sincosf(base * (float)u, &s, &c);
#pragma unroll
        for (int v = 0; v < NFREQ; ++v) {
            float drv = dr[u * NFREQ + v];
            float div = di[u * NFREQ + v];
            re[v] += drv * c - div * s;
            im[v] += drv * s + div * c;
        }
    }
    const float f = 1.0f / 1024.0f;   // 2*SCALING/sqrt(OUT*IN)
    Ac[r * NFREQ + 0] = 0.5f * f * re[0];
    Bc[r * NFREQ + 0] = 0.f;
#pragma unroll
    for (int v = 1; v < NFREQ; ++v) {
        Ac[r * NFREQ + v] =  f * re[v];
        Bc[r * NFREQ + v] = -f * im[v];
    }
}

// ---------------------------------------------------------------------------
// Kernel B: one block per output row. Build W_mod = W + C, reduce row norm,
// write bf16 W_dora = m * W_mod / (norm + eps), row-major [OUT, IN] = [N, K].
// ---------------------------------------------------------------------------
__global__ __launch_bounds__(256)
void dora_row(const float* __restrict__ W, const float* __restrict__ m,
              const float* __restrict__ Ac, const float* __restrict__ Bc,
              bf16* __restrict__ Wb)
{
    __shared__ float rowbuf[IN_DIM];
    __shared__ float red[256];
    __shared__ float cA[NFREQ], cB[NFREQ];
    __shared__ float sScale;

    const int r = blockIdx.x;
    const int tid = threadIdx.x;
    if (tid < NFREQ) { cA[tid] = Ac[r * NFREQ + tid]; cB[tid] = Bc[r * NFREQ + tid]; }
    __syncthreads();

    float a[NFREQ], b[NFREQ];
#pragma unroll
    for (int v = 0; v < NFREQ; ++v) { a[v] = cA[v]; b[v] = cB[v]; }

    float ss = 0.f;
    for (int c = tid; c < IN_DIM; c += 256) {
        float s1, c1;
        __sincosf((TWO_PI_F / (float)IN_DIM) * (float)c, &s1, &c1);
        // rotation recurrence for cos/sin(v*phi)
        float cv = c1, sv = s1;
        float corr = a[0] + a[1] * cv + b[1] * sv;
#pragma unroll
        for (int v = 2; v < NFREQ; ++v) {
            float cn = cv * c1 - sv * s1;
            float sn = sv * c1 + cv * s1;
            cv = cn; sv = sn;
            corr += a[v] * cv + b[v] * sv;
        }
        float w = W[(size_t)r * IN_DIM + c] + corr;
        rowbuf[c] = w;
        ss += w * w;
    }
    red[tid] = ss;
    __syncthreads();
#pragma unroll
    for (int ofs = 128; ofs > 0; ofs >>= 1) {
        if (tid < ofs) red[tid] += red[tid + ofs];
        __syncthreads();
    }
    if (tid == 0) sScale = m[r] / (sqrtf(red[0]) + EPS_F);
    __syncthreads();
    const float sc = sScale;
    for (int c = tid; c < IN_DIM; c += 256)
        Wb[(size_t)r * IN_DIM + c] = (bf16)(rowbuf[c] * sc);
}

// ---------------------------------------------------------------------------
// Kernel C: x f32 -> bf16 (4 elements per thread).
// ---------------------------------------------------------------------------
__global__ __launch_bounds__(256)
void convert_x(const float* __restrict__ x, bf16* __restrict__ Xb)
{
    size_t i4 = (size_t)(blockIdx.x) * 256 + threadIdx.x;
    size_t base = i4 * 4;
    float4 f = *(const float4*)(x + base);
    v4bf o;
    o[0] = (bf16)f.x; o[1] = (bf16)f.y; o[2] = (bf16)f.z; o[3] = (bf16)f.w;
    *(v4bf*)(Xb + base) = o;
}

// ---------------------------------------------------------------------------
// TDM: issue one 2D tile load (rows x 32 bf16) global -> LDS.
// D# pad: +16B after every 64B row  =>  LDS row stride 80B (bank-spread).
// 6-arg builtin variant (clang-23 / therock headers): groups 0..1 live,
// groups 2/3 + extra group zero (2D tensor, tile_dim2 = 0).
// ---------------------------------------------------------------------------
#define BK          32
#define BM          128
#define BN          128
#define LDS_STRIDE  40        // bf16 elements per LDS row (80 bytes)

__device__ __forceinline__ void tdm_load_tile(const bf16* gbase, uint32_t lds_byte_addr,
                                              uint32_t row_start, uint32_t k_start,
                                              uint32_t rows_total)
{
    uint64_t ga = (uint64_t)(uintptr_t)gbase
                + ((uint64_t)row_start * IN_DIM + (uint64_t)k_start) * 2ull;
    u32x4 g0;
    g0[0] = 1u;                                    // count = 1 valid descriptor
    g0[1] = lds_byte_addr;                         // LDS destination (bytes)
    g0[2] = (uint32_t)ga;                          // global addr [31:0]
    g0[3] = (uint32_t)(ga >> 32) | (2u << 30);     // addr [56:32] | type=2 (image)
    i32x8 g1;
    // data_size=1 (2B) | pad_enable | pad_interval=16DW(code3) | pad_amount=4DW(code3)
    g1[0] = (1 << 16) | (1 << 20) | (3 << 22) | (3 << 25);
    g1[1] = (int)((IN_DIM & 0xFFFF) << 16);                         // tensor_dim0 lo16
    g1[2] = (int)((IN_DIM >> 16) | ((rows_total & 0xFFFF) << 16));  // dim0 hi | dim1 lo
    g1[3] = (int)((rows_total >> 16) | (32u << 16));                // dim1 hi | tile_dim0=32
    g1[4] = BM;                                    // tile_dim1=128, tile_dim2=0
    g1[5] = IN_DIM;                                // tensor_dim0_stride lo32
    g1[6] = 0;                                     // stride hi | dim1_stride lo (2D: unused)
    g1[7] = 0;
    i32x4 z4 = {0, 0, 0, 0};
    i32x8 z8 = {0, 0, 0, 0, 0, 0, 0, 0};
    __builtin_amdgcn_tensor_load_to_lds(g0, g1, z4, z4, z8, 0);
}

// ---------------------------------------------------------------------------
// Fragment loads from padded LDS tiles (row-major [rows][K=32], stride 40 elems).
// A frag (16x32 bf16): lane L -> row m0 + (L%16), K = {kb..kb+7, kb+16..kb+23},
//                      kb = (L<16) ? 0 : 8.
// B frag (stored as B^T rows [N][K]): lane L -> row n0 + (L%16),
//                      K = {kb..kb+7, kb+16..kb+23}, kb = (L<16) ? 0 : 8.
// ---------------------------------------------------------------------------
__device__ __forceinline__ v16bf load_a_frag(const bf16* tile, int m0, int lane)
{
    int r  = m0 + (lane & 15);
    int kb = (lane < 16) ? 0 : 8;
    const bf16* p = tile + r * LDS_STRIDE + kb;
    v8bf lo = *(const v8bf*)p;
    v8bf hi = *(const v8bf*)(p + 16);
    v16bf f;
#pragma unroll
    for (int i = 0; i < 8; ++i) { f[i] = lo[i]; f[i + 8] = hi[i]; }
    return f;
}

__device__ __forceinline__ v16bf load_b_frag(const bf16* tile, int n0, int lane)
{
    int r  = n0 + (lane & 15);
    int kb = (lane < 16) ? 0 : 8;
    const bf16* p = tile + r * LDS_STRIDE + kb;
    v8bf lo = *(const v8bf*)p;
    v8bf hi = *(const v8bf*)(p + 16);
    v16bf f;
#pragma unroll
    for (int i = 0; i < 8; ++i) { f[i] = lo[i]; f[i + 8] = hi[i]; }
    return f;
}

// ---------------------------------------------------------------------------
// Kernel D: GEMM  out[M,N] = Xb[M,K] * Wb[N,K]^T + bias.
// 128x128 block tile, 8 waves, each wave a 32x64 sub-tile (2x4 WMMA accums).
// TDM double-buffered K pipeline, s_wait_tensorcnt + barrier sync.
// ---------------------------------------------------------------------------
__global__ __launch_bounds__(256)
void gemm_bf16(const bf16* __restrict__ Xb, const bf16* __restrict__ Wb,
               const float* __restrict__ bias, float* __restrict__ out)
{
    __shared__ __align__(16) bf16 sX[2][BM * LDS_STRIDE];
    __shared__ __align__(16) bf16 sW[2][BN * LDS_STRIDE];

    const int tid  = threadIdx.x;
    const int lane = tid & 31;
    const int wave = tid >> 5;
    const int mTile = blockIdx.y * BM;
    const int nTile = blockIdx.x * BN;
    const int wm = (wave >> 1) * 32;   // 0,32,64,96
    const int wn = (wave & 1) * 64;    // 0,64

    v8f acc[2][4];
#pragma unroll
    for (int i = 0; i < 2; ++i)
#pragma unroll
        for (int j = 0; j < 4; ++j)
#pragma unroll
            for (int g = 0; g < 8; ++g) acc[i][j][g] = 0.f;

    const int T = IN_DIM / BK;   // 128 K-steps

    if (tid < 32) {
        tdm_load_tile(Xb, (uint32_t)(uintptr_t)(void*)&sX[0][0], mTile, 0, MROWS);
        tdm_load_tile(Wb, (uint32_t)(uintptr_t)(void*)&sW[0][0], nTile, 0, OUT_DIM);
    }

    for (int t = 0; t < T; ++t) {
        const int cur = t & 1;
        if (tid < 32) {
            if (t + 1 < T) {
                tdm_load_tile(Xb, (uint32_t)(uintptr_t)(void*)&sX[cur ^ 1][0],
                              mTile, (t + 1) * BK, MROWS);
                tdm_load_tile(Wb, (uint32_t)(uintptr_t)(void*)&sW[cur ^ 1][0],
                              nTile, (t + 1) * BK, OUT_DIM);
                __builtin_amdgcn_s_wait_tensorcnt(2);   // current buffer done
            } else {
                __builtin_amdgcn_s_wait_tensorcnt(0);   // drain
            }
        }
        __syncthreads();

        v16bf a0 = load_a_frag(&sX[cur][0], wm, lane);
        v16bf a1 = load_a_frag(&sX[cur][0], wm + 16, lane);
        v16bf b0 = load_b_frag(&sW[cur][0], wn +  0, lane);
        v16bf b1 = load_b_frag(&sW[cur][0], wn + 16, lane);
        v16bf b2 = load_b_frag(&sW[cur][0], wn + 32, lane);
        v16bf b3 = load_b_frag(&sW[cur][0], wn + 48, lane);

        acc[0][0] = __builtin_amdgcn_wmma_f32_16x16x32_bf16(false, a0, false, b0, (short)0, acc[0][0], false, false);
        acc[0][1] = __builtin_amdgcn_wmma_f32_16x16x32_bf16(false, a0, false, b1, (short)0, acc[0][1], false, false);
        acc[0][2] = __builtin_amdgcn_wmma_f32_16x16x32_bf16(false, a0, false, b2, (short)0, acc[0][2], false, false);
        acc[0][3] = __builtin_amdgcn_wmma_f32_16x16x32_bf16(false, a0, false, b3, (short)0, acc[0][3], false, false);
        acc[1][0] = __builtin_amdgcn_wmma_f32_16x16x32_bf16(false, a1, false, b0, (short)0, acc[1][0], false, false);
        acc[1][1] = __builtin_amdgcn_wmma_f32_16x16x32_bf16(false, a1, false, b1, (short)0, acc[1][1], false, false);
        acc[1][2] = __builtin_amdgcn_wmma_f32_16x16x32_bf16(false, a1, false, b2, (short)0, acc[1][2], false, false);
        acc[1][3] = __builtin_amdgcn_wmma_f32_16x16x32_bf16(false, a1, false, b3, (short)0, acc[1][3], false, false);

        __syncthreads();
    }

    // Epilogue. C/D layout: VGPR g, lanes 0-15 -> M = g, lanes 16-31 -> M = 8+g,
    // N = lane % 16 within each 16x16 tile.
    const int nl = lane & 15;
    const int mofs = (lane < 16) ? 0 : 8;
#pragma unroll
    for (int j = 0; j < 4; ++j) {
        const int ncol = nTile + wn + j * 16 + nl;
        const float bv = bias[ncol];
#pragma unroll
        for (int i = 0; i < 2; ++i) {
            const int mbase = mTile + wm + i * 16 + mofs;
#pragma unroll
            for (int g = 0; g < 8; ++g)
                out[(size_t)(mbase + g) * OUT_DIM + ncol] = acc[i][j][g] + bv;
        }
    }
}

// ---------------------------------------------------------------------------
// Host-side launcher. Inputs: x, W, bias, delta_real, delta_imag, m (all f32).
// ---------------------------------------------------------------------------
extern "C" void kernel_launch(void* const* d_in, const int* in_sizes, int n_in,
                              void* d_out, int out_size, void* d_ws, size_t ws_size,
                              hipStream_t stream)
{
    const float* x    = (const float*)d_in[0];
    const float* W    = (const float*)d_in[1];
    const float* bias = (const float*)d_in[2];
    const float* dr   = (const float*)d_in[3];
    const float* di   = (const float*)d_in[4];
    const float* m    = (const float*)d_in[5];
    float* out = (float*)d_out;

    char* ws = (char*)d_ws;
    bf16*  Xb = (bf16*)ws;                                           // 67.1 MB
    bf16*  Wb = (bf16*)(ws + (size_t)MROWS * IN_DIM * 2);            // 33.6 MB
    float* Ac = (float*)(ws + (size_t)MROWS * IN_DIM * 2
                            + (size_t)OUT_DIM * IN_DIM * 2);
    float* Bc = Ac + (size_t)OUT_DIM * NFREQ;

    fou_coeffs<<<OUT_DIM / 256, 256, 0, stream>>>(dr, di, Ac, Bc);
    dora_row<<<OUT_DIM, 256, 0, stream>>>(W, m, Ac, Bc, Wb);
    convert_x<<<(int)(((size_t)MROWS * IN_DIM / 4) / 256), 256, 0, stream>>>(x, Xb);

    dim3 grid(OUT_DIM / BN, MROWS / BM);   // (32, 64)
    gemm_bf16<<<grid, 256, 0, stream>>>(Xb, Wb, bias, out);
}
